// DimeNet_79276506350113
// MI455X (gfx1250) — matrix operand order repaired and checked by reading, not compiled, not run on hardware
//
#include <hip/hip_runtime.h>
#include <math.h>

// ---------------------------------------------------------------------------
// Types for CDNA5 WMMA (wave32, 16x16x32 bf16 -> f32)
// ---------------------------------------------------------------------------
typedef __attribute__((ext_vector_type(16))) __bf16 v16bf;
typedef __attribute__((ext_vector_type(8)))  float  v8f;

#define NSPH 7
#define NRAD 6
#define HD   128

// Spherical Bessel j_l zeros, l=0..6, first 6 roots each (module constants in ref)
__constant__ float c_zeros[42] = {
  3.14159265f,  6.28318531f,  9.42477796f, 12.56637061f, 15.70796327f, 18.84955592f,
  4.49340946f,  7.72525184f, 10.90412166f, 14.06619391f, 17.22075527f, 20.37130296f,
  5.76345920f,  9.09501133f, 12.32294097f, 15.51460301f, 18.68903636f, 21.85387422f,
  6.98793200f, 10.41711855f, 13.69802315f, 16.92362129f, 20.12180617f, 23.30424699f,
  8.18256145f, 11.70490715f, 15.03966471f, 18.30125596f, 21.52541773f, 24.72756555f,
  9.35581211f, 12.96653017f, 16.35470964f, 19.65315210f, 22.90455065f, 26.12775014f,
 10.51283541f, 14.20739246f, 17.64797487f, 20.98346307f, 24.26276804f, 27.50786839f };

// ---------------------------------------------------------------------------
// Helpers
// ---------------------------------------------------------------------------
__device__ __forceinline__ unsigned short f2bf(float f) {
  unsigned int u = __float_as_uint(f);
  u = (u + 0x7FFFu + ((u >> 16) & 1u)) >> 16;   // round-to-nearest-even
  return (unsigned short)u;
}
__device__ __forceinline__ float silu(float v) {
  return v * (1.0f / (1.0f + __expf(-v)));
}
__device__ __forceinline__ float sph_jn(int l, float x) {
  float s, c;
  sincosf(x, &s, &c);
  float j0 = s / x;
  if (l == 0) return j0;
  float jm = j0, jc = s / (x * x) - c / x;
  for (int n = 1; n < l; ++n) { float jn = (2.0f * n + 1.0f) / x * jc - jm; jm = jc; jc = jn; }
  return jc;
}

// ---------------------------------------------------------------------------
// Weight prepack: f32 KxN (row-major) -> bf16 packed per-lane WMMA B layout
// layout: [ntile][ktile][lane(32)][vgpr(8)] u32, each u32 = {K, K+1} bf16 pair
// mode 0: normal row-major KxN.  mode 1: W (H,NBIL,H) -> Wcat[k=hin][n=l*128+hout]
// ---------------------------------------------------------------------------
__global__ void pack_b_kernel(const float* __restrict__ W, unsigned int* __restrict__ out,
                              int K, int N, int mode) {
  int idx = blockIdx.x * blockDim.x + threadIdx.x;
  int Kt = K >> 5;
  int total = (N >> 4) * Kt * 256;
  if (idx >= total) return;
  int p    = idx & 7;
  int lane = (idx >> 3) & 31;
  int tile = idx >> 8;
  int kt = tile % Kt, nt = tile / Kt;
  int n = nt * 16 + (lane & 15);
  int k = kt * 32 + ((lane >> 4) << 3) + (p << 1) + ((p >> 2) << 3);   // lanes 0-15: K 0..7,16..23; 16-31: K 8..15,24..31
  float f0, f1;
  if (mode == 0) {
    f0 = W[(size_t)k * N + n];
    f1 = W[(size_t)(k + 1) * N + n];
  } else {        // element(k,n) = Wp[(n&127)*1024 + (n>>7)*128 + k]
    size_t base = (size_t)(n & 127) * 1024 + (size_t)(n >> 7) * 128;
    f0 = W[base + k];
    f1 = W[base + k + 1];
  }
  out[idx] = (unsigned int)f2bf(f0) | ((unsigned int)f2bf(f1) << 16);
}

// ---------------------------------------------------------------------------
// WMMA GEMM: C[MxN] = epilogue(A[MxK]@B + bias)
// one wave per block; wave computes 32(M) x 64(N): MT=2 A fragments x NT=4
// B subtiles -> 8 v_wmma per 32-K step, 16 b128 loads.
// A fetched as 4x float4 per fragment (two contiguous 8-float runs),
// converted f32->bf16 with native casts.
// mode: 0 none, 1 silu, 2 silu then *extra, 3 silu then +extra
// ---------------------------------------------------------------------------
__global__ __launch_bounds__(32) void
gemm_wmma(const float* __restrict__ A, const unsigned int* __restrict__ Bp,
          const float* __restrict__ bias, const float* __restrict__ extra,
          float* __restrict__ C, int M, int K, int N, int mode) {
  int lane = threadIdx.x;
  int tm   = blockIdx.x;
  int nt0  = blockIdx.y << 2;
  int kh   = lane >> 4;
  int Kt   = K >> 5;
  int row0 = tm * 32 + (lane & 15);
  int row1 = row0 + 16;
  bool rok0 = row0 < M;
  bool rok1 = row1 < M;
  const float* arow0 = A + (size_t)row0 * K;
  const float* arow1 = A + (size_t)row1 * K;

  v8f acc[2][4];
#pragma unroll
  for (int t = 0; t < 2; ++t)
#pragma unroll
    for (int s = 0; s < 4; ++s) acc[t][s] = (v8f){};

  for (int kt = 0; kt < Kt; ++kt) {
    int kb = kt * 32 + kh * 8;   // A layout: lanes0-15 K in {kb..kb+7, kb+16..kb+23}
    union { v16bf v; unsigned int u[8]; } a0, a1;
    {
      float4 z = make_float4(0.f, 0.f, 0.f, 0.f);
      float4 q0 = z, q1 = z, q2 = z, q3 = z;
      if (rok0) {
        q0 = *(const float4*)(arow0 + kb);
        q1 = *(const float4*)(arow0 + kb + 4);
        q2 = *(const float4*)(arow0 + kb + 16);
        q3 = *(const float4*)(arow0 + kb + 20);
      }
      a0.v[0]  = (__bf16)q0.x; a0.v[1]  = (__bf16)q0.y;
      a0.v[2]  = (__bf16)q0.z; a0.v[3]  = (__bf16)q0.w;
      a0.v[4]  = (__bf16)q1.x; a0.v[5]  = (__bf16)q1.y;
      a0.v[6]  = (__bf16)q1.z; a0.v[7]  = (__bf16)q1.w;
      a0.v[8]  = (__bf16)q2.x; a0.v[9]  = (__bf16)q2.y;
      a0.v[10] = (__bf16)q2.z; a0.v[11] = (__bf16)q2.w;
      a0.v[12] = (__bf16)q3.x; a0.v[13] = (__bf16)q3.y;
      a0.v[14] = (__bf16)q3.z; a0.v[15] = (__bf16)q3.w;
      q0 = q1 = q2 = q3 = z;
      if (rok1) {
        q0 = *(const float4*)(arow1 + kb);
        q1 = *(const float4*)(arow1 + kb + 4);
        q2 = *(const float4*)(arow1 + kb + 16);
        q3 = *(const float4*)(arow1 + kb + 20);
      }
      a1.v[0]  = (__bf16)q0.x; a1.v[1]  = (__bf16)q0.y;
      a1.v[2]  = (__bf16)q0.z; a1.v[3]  = (__bf16)q0.w;
      a1.v[4]  = (__bf16)q1.x; a1.v[5]  = (__bf16)q1.y;
      a1.v[6]  = (__bf16)q1.z; a1.v[7]  = (__bf16)q1.w;
      a1.v[8]  = (__bf16)q2.x; a1.v[9]  = (__bf16)q2.y;
      a1.v[10] = (__bf16)q2.z; a1.v[11] = (__bf16)q2.w;
      a1.v[12] = (__bf16)q3.x; a1.v[13] = (__bf16)q3.y;
      a1.v[14] = (__bf16)q3.z; a1.v[15] = (__bf16)q3.w;
    }
#pragma unroll
    for (int s = 0; s < 4; ++s) {
      union { v16bf v; uint4 q[2]; } b;
      const uint4* bp = (const uint4*)(Bp + (((size_t)(nt0 + s) * Kt + kt) << 8) + (lane << 3));
      b.q[0] = bp[0];
      b.q[1] = bp[1];
      acc[0][s] = __builtin_amdgcn_wmma_f32_16x16x32_bf16(false, a0.v, false, b.v,
                                                          (short)0, acc[0][s], false, false);
      acc[1][s] = __builtin_amdgcn_wmma_f32_16x16x32_bf16(false, a1.v, false, b.v,
                                                          (short)0, acc[1][s], false, false);
    }
  }
#pragma unroll
  for (int s = 0; s < 4; ++s) {
    int n = (nt0 + s) * 16 + (lane & 15);
    float bv = bias ? bias[n] : 0.f;
#pragma unroll
    for (int t = 0; t < 2; ++t) {
#pragma unroll
      for (int r = 0; r < 8; ++r) {
        int m = tm * 32 + t * 16 + r + kh * 8;  // C layout: vgpr r: lanes0-15 M=r, 16-31 M=r+8
        if (m < M) {
          float v = acc[t][s][r] + bv;
          if (mode >= 1) v = silu(v);
          size_t o = (size_t)m * N + n;
          if (mode == 2)      v *= extra[o];
          else if (mode == 3) v += extra[o];
          C[o] = v;
        }
      }
    }
  }
}

// ---------------------------------------------------------------------------
// Small-K linear: C[m, cofs+n (ld=ldc)] = act(sum_k A[m,k]W[k,n] + b[n])
// ---------------------------------------------------------------------------
__global__ void small_lin(const float* __restrict__ A, const float* __restrict__ W,
                          const float* __restrict__ bias, float* __restrict__ C,
                          int M, int K, int N, int ldc, int cofs, int act) {
  int idx = blockIdx.x * blockDim.x + threadIdx.x;
  if (idx >= M * N) return;
  int m = idx / N, n = idx - m * N;
  float s = bias ? bias[n] : 0.f;
  for (int k = 0; k < K; ++k) s += A[(size_t)m * K + k] * W[(size_t)k * N + n];
  if (act) s = silu(s);
  C[(size_t)m * ldc + cofs + n] = s;
}

// ---------------------------------------------------------------------------
// Geometry
// ---------------------------------------------------------------------------
__global__ void init_bnorm(float* __restrict__ bnorm) {
  int t = threadIdx.x;
  if (t < 42) {
    int l = t / 6;
    bnorm[t] = sqrtf(2.0f) / fabsf(sph_jn(l + 1, c_zeros[t]));
  }
}

__global__ void edge_geom(const float* __restrict__ pos, const int* __restrict__ esrc,
                          const int* __restrict__ edst, const float* __restrict__ freq,
                          const float* __restrict__ bnorm, float* __restrict__ rbf,
                          float* __restrict__ sbfr, int E) {
  int e = blockIdx.x * blockDim.x + threadIdx.x;
  if (e >= E) return;
  int i = edst[e], j = esrc[e];
  float dx = pos[i * 3 + 0] - pos[j * 3 + 0];
  float dy = pos[i * 3 + 1] - pos[j * 3 + 1];
  float dz = pos[i * 3 + 2] - pos[j * 3 + 2];
  float d = sqrtf(dx * dx + dy * dy + dz * dz) * 0.2f;   // dist / CUTOFF
  float x5 = d * d * d * d * d;                          // ENVP=5 -> p=6
  float env = (d < 1.f) ? (1.f / d - 28.f * x5 + 48.f * x5 * d - 21.f * x5 * d * d) : 0.f;
#pragma unroll
  for (int r = 0; r < NRAD; ++r)
    rbf[(size_t)e * NRAD + r] = env * sinf(freq[r] * d);
  for (int l = 0; l < NSPH; ++l)
#pragma unroll
    for (int r = 0; r < NRAD; ++r)
      sbfr[(size_t)e * 42 + l * 6 + r] = env * bnorm[l * 6 + r] * sph_jn(l, d * c_zeros[l * 6 + r]);
}

__global__ void triplet_sbf(const float* __restrict__ pos, const int* __restrict__ ii,
                            const int* __restrict__ jj, const int* __restrict__ kk,
                            const int* __restrict__ ikj, const float* __restrict__ sbfr,
                            float* __restrict__ sbf, int T) {
  int t = blockIdx.x * blockDim.x + threadIdx.x;
  if (t >= T) return;
  int a0 = ii[t], a1 = jj[t], a2 = kk[t];
  float jx = pos[a1 * 3 + 0] - pos[a0 * 3 + 0];
  float jy = pos[a1 * 3 + 1] - pos[a0 * 3 + 1];
  float jz = pos[a1 * 3 + 2] - pos[a0 * 3 + 2];
  float kx = pos[a2 * 3 + 0] - pos[a0 * 3 + 0];
  float ky = pos[a2 * 3 + 1] - pos[a0 * 3 + 1];
  float kz = pos[a2 * 3 + 2] - pos[a0 * 3 + 2];
  float dp = jx * kx + jy * ky + jz * kz;
  float cx = jy * kz - jz * ky, cy = jz * kx - jx * kz, cz = jx * ky - jy * kx;
  float cb = sqrtf(cx * cx + cy * cy + cz * cz);
  float ct = cosf(atan2f(cb, dp));
  float leg[NSPH];
  leg[0] = 1.f; leg[1] = ct;
#pragma unroll
  for (int l = 2; l < NSPH; ++l)
    leg[l] = ((2.f * l - 1.f) * ct * leg[l - 1] - (l - 1.f) * leg[l - 2]) / (float)l;
  const float inv4pi = 0.07957747154594767f;
  int e = ikj[t];
  for (int l = 0; l < NSPH; ++l) {
    float cbf = leg[l] * sqrtf((2.f * l + 1.f) * inv4pi);
#pragma unroll
    for (int r = 0; r < NRAD; ++r)
      sbf[(size_t)t * 42 + l * 6 + r] = sbfr[(size_t)e * 42 + l * 6 + r] * cbf;
  }
}

// ---------------------------------------------------------------------------
// Irregular graph ops
// ---------------------------------------------------------------------------
__global__ void concat_emb(float* __restrict__ cat, const float* __restrict__ emb,
                           const int* __restrict__ z, const int* __restrict__ edst,
                           const int* __restrict__ esrc, int E) {
  int idx = blockIdx.x * blockDim.x + threadIdx.x;
  if (idx >= E * HD) return;
  int e = idx >> 7, h = idx & 127;
  cat[(size_t)e * 384 + h]       = emb[(size_t)z[edst[e]] * HD + h];
  cat[(size_t)e * 384 + 128 + h] = emb[(size_t)z[esrc[e]] * HD + h];
}

__global__ void triplet_msg(const float* __restrict__ sbfl, const float* __restrict__ Abig,
                            const int* __restrict__ ikj, const int* __restrict__ iji,
                            float* __restrict__ outsum, int T) {
  int t = blockIdx.x;
  if (t >= T) return;
  __shared__ float sl[8];
  int h = threadIdx.x;
  if (h < 8) sl[h] = sbfl[(size_t)t * 8 + h];
  __syncthreads();
  int kj = ikj[t], ji = iji[t];
  const float* arow = Abig + (size_t)kj * 1024 + h;
  float s = 0.f;
#pragma unroll
  for (int l = 0; l < 8; ++l) s += sl[l] * arow[l * HD];
  atomicAdd(&outsum[(size_t)ji * HD + h], s);
}

__global__ void vadd(const float* __restrict__ a, const float* __restrict__ b,
                     float* __restrict__ c, size_t n) {
  size_t i = (size_t)blockIdx.x * blockDim.x + threadIdx.x;
  if (i < n) c[i] = a[i] + b[i];
}

__global__ void edge_scatter(const float* __restrict__ rbfl, const float* __restrict__ x,
                             const int* __restrict__ edst, float* __restrict__ ynode, int E) {
  int idx = blockIdx.x * blockDim.x + threadIdx.x;
  if (idx >= E * HD) return;
  int e = idx >> 7, h = idx & 127;
  atomicAdd(&ynode[(size_t)edst[e] * HD + h], rbfl[idx] * x[idx]);
}

__global__ void node_dot(const float* __restrict__ y, const float* __restrict__ w,
                         float* __restrict__ out, int N) {
  int n = blockIdx.x * blockDim.x + threadIdx.x;
  if (n >= N) return;
  float s = 0.f;
#pragma unroll 8
  for (int h = 0; h < HD; ++h) s += y[(size_t)n * HD + h] * w[h];
  atomicAdd(out, s);
}

// ---------------------------------------------------------------------------
// Host orchestration
// ---------------------------------------------------------------------------
extern "C" void kernel_launch(void* const* d_in, const int* in_sizes, int n_in,
                              void* d_out, int out_size, void* d_ws, size_t ws_size,
                              hipStream_t stream) {
  (void)n_in; (void)ws_size;
  auto F = [&](int i) { return (const float*)d_in[i]; };
  auto I = [&](int i) { return (const int*)d_in[i]; };

  const int* z    = I(0);
  const float* pos = F(1);
  const int* esrc = I(2);
  const int* edst = I(3);
  const int* ti   = I(4);
  const int* tj   = I(5);
  const int* tk   = I(6);
  const int* tkj  = I(7);
  const int* tji  = I(8);
  // params (jax tree_leaves order: sorted dict keys, lists in order)
  // 9: emb_lin.b  10: emb_lin.w  11: emb_lin_rbf.b  12: emb_lin_rbf.w  13: emb_table  14: freq
  auto IBX = [&](int b, int o) { return 15 + b * 21 + o; };
  // per int block: 0:W 1:a0.b1 2:a0.b2 3:a0.w1 4:a0.w2 5:a1.b1 6:a1.b2 7:a1.w1 8:a1.w2
  //                9:be.b1 10:be.b2 11:be.w1 12:be.w2 13:lin.b 14:lin.w
  //                15:lin_ji.b 16:lin_ji.w 17:lin_kj.b 18:lin_kj.w 19:lin_rbf 20:lin_sbf
  auto OBX = [&](int b, int o) { return 78 + b * 8 + o; };
  // per out block: 0:lin_out 1:lin_rbf 2:l0.b 3:l0.w 4:l1.b 5:l1.w 6:l2.b 7:l2.w

  const int N = in_sizes[1] / 3;
  const int E = in_sizes[2];
  const int T = in_sizes[4];

  // ---- workspace carve-up (deterministic) ----
  char* ws = (char*)d_ws;
  size_t off = 0;
  auto alloc = [&](size_t nbytes) -> void* {
    off = (off + 255) & ~(size_t)255;
    void* p = ws + off;
    off += nbytes;
    return p;
  };
  float* bnorm = (float*)alloc(42 * sizeof(float));
  float* rbf   = (float*)alloc((size_t)E * 6 * sizeof(float));
  float* sbfr  = (float*)alloc((size_t)E * 42 * sizeof(float));
  float* sbf   = (float*)alloc((size_t)T * 42 * sizeof(float));
  float* sbfl  = (float*)alloc((size_t)T * 8 * sizeof(float));
  float* rbfl  = (float*)alloc((size_t)E * HD * sizeof(float));
  float* xbuf  = (float*)alloc((size_t)E * HD * sizeof(float));
  float* bufA  = (float*)alloc((size_t)E * HD * sizeof(float));
  float* bufB  = (float*)alloc((size_t)E * HD * sizeof(float));
  float* bufC  = (float*)alloc((size_t)E * HD * sizeof(float));
  float* bufD  = (float*)alloc((size_t)E * HD * sizeof(float));
  float* Abig  = (float*)alloc((size_t)E * 1024 * sizeof(float));
  float* cat   = Abig;   // E x 384 fits inside E x 1024; disjoint in time
  float* ynode = (float*)alloc((size_t)N * HD * sizeof(float));
  float* ytmp  = (float*)alloc((size_t)N * HD * sizeof(float));

  auto packmat = [&](const float* W, int K, int Ncols, int mode) -> const unsigned int* {
    unsigned int* dst = (unsigned int*)alloc((size_t)K * Ncols * 2);
    int total = (Ncols / 16) * (K / 32) * 256;
    pack_b_kernel<<<(total + 255) / 256, 256, 0, stream>>>(W, dst, K, Ncols, mode);
    return dst;
  };
  auto gemm = [&](const float* A, const unsigned int* Bp, const float* bias,
                  const float* extra, float* C, int M, int K, int Nc, int mode) {
    dim3 g((M + 31) / 32, Nc / 64);
    gemm_wmma<<<g, 32, 0, stream>>>(A, Bp, bias, extra, C, M, K, Nc, mode);
  };
  auto slin = [&](const float* A, const float* W, const float* bias, float* C,
                  int M, int K, int Nc, int ldc, int cofs, int act) {
    small_lin<<<((long)M * Nc + 255) / 256, 256, 0, stream>>>(A, W, bias, C, M, K, Nc, ldc, cofs, act);
  };

  // ---- pack all bf16 weight operands once per launch ----
  const unsigned int* pk_emb = packmat(F(10), 384, 128, 0);
  const unsigned int *pk_ji[3], *pk_kj[3], *pk_wcat[3], *pk_bw1[3], *pk_bw2[3], *pk_lin[3], *pk_aw[3][4];
  for (int b = 0; b < 3; ++b) {
    pk_ji[b]    = packmat(F(IBX(b, 16)), 128, 128, 0);
    pk_kj[b]    = packmat(F(IBX(b, 18)), 128, 128, 0);
    pk_wcat[b]  = packmat(F(IBX(b, 0)),  128, 1024, 1);
    pk_bw1[b]   = packmat(F(IBX(b, 11)), 128, 128, 0);
    pk_bw2[b]   = packmat(F(IBX(b, 12)), 128, 128, 0);
    pk_lin[b]   = packmat(F(IBX(b, 14)), 128, 128, 0);
    pk_aw[b][0] = packmat(F(IBX(b, 3)),  128, 128, 0);
    pk_aw[b][1] = packmat(F(IBX(b, 4)),  128, 128, 0);
    pk_aw[b][2] = packmat(F(IBX(b, 7)),  128, 128, 0);
    pk_aw[b][3] = packmat(F(IBX(b, 8)),  128, 128, 0);
  }
  const unsigned int* pk_out[4][3];
  for (int ob = 0; ob < 4; ++ob)
    for (int li = 0; li < 3; ++li)
      pk_out[ob][li] = packmat(F(OBX(ob, 3 + 2 * li)), 128, 128, 0);

  // ---- geometry ----
  hipMemsetAsync(d_out, 0, (size_t)out_size * sizeof(float), stream);
  init_bnorm<<<1, 64, 0, stream>>>(bnorm);
  edge_geom<<<(E + 127) / 128, 128, 0, stream>>>(pos, esrc, edst, F(14), bnorm, rbf, sbfr, E);
  triplet_sbf<<<(T + 127) / 128, 128, 0, stream>>>(pos, ti, tj, tk, tkj, sbfr, sbf, T);

  // ---- embedding block: x = silu([h_i | h_j | silu(rbf@Wr+br)] @ W + b) ----
  slin(rbf, F(12), F(11), cat, E, 6, 128, 384, 256, 1);       // rbf_h -> cat[:,256:384]
  concat_emb<<<((long)E * HD + 255) / 256, 256, 0, stream>>>(cat, F(13), z, edst, esrc, E);
  gemm(cat, pk_emb, F(9), nullptr, xbuf, E, 384, 128, 1);

  float* xcur = xbuf;
  float* xalt = bufB;

  auto outblock = [&](int ob, const float* xin) {
    slin(rbf, F(OBX(ob, 1)), nullptr, rbfl, E, 6, 128, 128, 0, 0);
    hipMemsetAsync(ynode, 0, (size_t)N * HD * sizeof(float), stream);
    edge_scatter<<<((long)E * HD + 255) / 256, 256, 0, stream>>>(rbfl, xin, edst, ynode, E);
    gemm(ynode, pk_out[ob][0], F(OBX(ob, 2)), nullptr, ytmp,  N, 128, 128, 1);
    gemm(ytmp,  pk_out[ob][1], F(OBX(ob, 4)), nullptr, ynode, N, 128, 128, 1);
    gemm(ynode, pk_out[ob][2], F(OBX(ob, 6)), nullptr, ytmp,  N, 128, 128, 1);
    node_dot<<<(N + 127) / 128, 128, 0, stream>>>(ytmp, F(OBX(ob, 0)), (float*)d_out, N);
  };

  outblock(0, xcur);

  for (int b = 0; b < 3; ++b) {
    // rbf / sbf projections
    slin(rbf, F(IBX(b, 19)), nullptr, rbfl, E, 6, 128, 128, 0, 0);
    slin(sbf, F(IBX(b, 20)), nullptr, sbfl, T, 42, 8, 8, 0, 0);
    // x_ji, x_kj_pre
    gemm(xcur, pk_ji[b], F(IBX(b, 15)), nullptr, bufA, E, 128, 128, 1);          // x_ji
    gemm(xcur, pk_kj[b], F(IBX(b, 17)), rbfl,    xalt, E, 128, 128, 2);          // silu(..)*rbf_l
    // factorized bilinear: A[e, l*128+hout] = x_kj_pre @ Wcat
    gemm(xalt, pk_wcat[b], nullptr, nullptr, Abig, E, 128, 1024, 0);
    // triplet gather + segment-sum into edges
    hipMemsetAsync(bufC, 0, (size_t)E * HD * sizeof(float), stream);
    triplet_msg<<<T, 128, 0, stream>>>(sbfl, Abig, tkj, tji, bufC, T);
    // hh = x_ji + x_kj
    vadd<<<((long)E * HD + 255) / 256, 256, 0, stream>>>(bufA, bufC, bufA, (size_t)E * HD);
    // residual (before)
    gemm(bufA, pk_bw1[b], F(IBX(b, 9)),  nullptr, bufD, E, 128, 128, 1);
    gemm(bufD, pk_bw2[b], F(IBX(b, 10)), bufA,    bufA, E, 128, 128, 3);
    // hh = silu(hh@lin + b) + x   (written to xalt, which becomes next x)
    gemm(bufA, pk_lin[b], F(IBX(b, 13)), xcur, xalt, E, 128, 128, 3);
    // residuals (after) x2, in place on xalt
    gemm(xalt, pk_aw[b][0], F(IBX(b, 1)), nullptr, bufD, E, 128, 128, 1);
    gemm(bufD, pk_aw[b][1], F(IBX(b, 2)), xalt,    xalt, E, 128, 128, 3);
    gemm(xalt, pk_aw[b][2], F(IBX(b, 5)), nullptr, bufD, E, 128, 128, 1);
    gemm(bufD, pk_aw[b][3], F(IBX(b, 6)), xalt,    xalt, E, 128, 128, 3);
    // swap: xalt becomes current x
    float* t2 = xcur; xcur = xalt; xalt = t2;
    outblock(b + 1, xcur);
  }
}
